// MemoryAsLayerTitan_53455162966284
// MI455X (gfx1250) — matrix operand order repaired and checked by reading, not compiled, not run on hardware
//
#include <hip/hip_runtime.h>

// ---------------------------------------------------------------------------
// Problem constants (match the reference)
// ---------------------------------------------------------------------------
constexpr int Bc   = 2;
constexpr int Lc   = 2048;
constexpr int Dc   = 512;
constexpr int Pc   = 32;
constexpr int Mc   = 1024;
constexpr int Hc   = 8;
constexpr int HDc  = 64;
constexpr int WINc = 256;
constexpr int Sc   = Pc + Lc;    // 2080
constexpr int BSc  = Bc * Sc;    // 4160

typedef __bf16 bf16;
typedef bf16  v8bf  __attribute__((ext_vector_type(8)));
typedef bf16  v16bf __attribute__((ext_vector_type(16)));
typedef float v8f   __attribute__((ext_vector_type(8)));

#define DEV static __device__ __forceinline__

DEV v16bf cat8(v8bf lo, v8bf hi) {
  v16bf r;
#pragma unroll
  for (int t = 0; t < 8; ++t) { r[t] = lo[t]; r[8 + t] = hi[t]; }
  return r;
}

DEV v16bf z16() {
  v16bf r;
#pragma unroll
  for (int t = 0; t < 16; ++t) r[t] = (bf16)0.0f;
  return r;
}

DEV v8bf z8() {
  v8bf r;
#pragma unroll
  for (int t = 0; t < 8; ++t) r[t] = (bf16)0.0f;
  return r;
}

DEV v8f wmma_bf16(v16bf a, v16bf b, v8f c) {
  // D = A(16x32 bf16) * B(32x16 bf16) + C(16x16 f32)
  return __builtin_amdgcn_wmma_f32_16x16x32_bf16(
      /*neg_a=*/false, a, /*neg_b=*/false, b,
      /*c_mod=*/(short)0, c, /*reuse_a=*/false, /*reuse_b=*/false);
}

// ---------------------------------------------------------------------------
// Elementwise kernels
// ---------------------------------------------------------------------------
__global__ void f32_to_bf16_k(const float* __restrict__ in,
                              bf16* __restrict__ out, int n) {
  int i = blockIdx.x * blockDim.x + threadIdx.x;
  if (i < n) out[i] = (bf16)in[i];
}

__global__ void build_combined_k(const float* __restrict__ x,
                                 const float* __restrict__ pm,
                                 bf16* __restrict__ out) {
  int i = blockIdx.x * blockDim.x + threadIdx.x;
  if (i >= BSc * Dc) return;
  int d  = i % Dc;
  int sb = i / Dc;
  int s  = sb % Sc;
  int b  = sb / Sc;
  float v = (s < Pc) ? pm[s * Dc + d]
                     : x[((long)b * Lc + (s - Pc)) * Dc + d];
  out[i] = (bf16)v;
}

__global__ __launch_bounds__(256)
void softmax_rows_k(const float* __restrict__ in, bf16* __restrict__ out,
                    int ncols) {
  __shared__ float red[256];
  const int row = blockIdx.x;
  const float* p = in + (long)row * ncols;
  bf16* o = out + (long)row * ncols;
  const int tid = threadIdx.x;
  float mx = -3.0e38f;
  for (int c = tid; c < ncols; c += 256) mx = fmaxf(mx, p[c]);
  red[tid] = mx; __syncthreads();
  for (int s = 128; s > 0; s >>= 1) {
    if (tid < s) red[tid] = fmaxf(red[tid], red[tid + s]);
    __syncthreads();
  }
  mx = red[0]; __syncthreads();
  float sum = 0.f;
  for (int c = tid; c < ncols; c += 256) sum += __expf(p[c] - mx);
  red[tid] = sum; __syncthreads();
  for (int s = 128; s > 0; s >>= 1) {
    if (tid < s) red[tid] += red[tid + s];
    __syncthreads();
  }
  float inv = 1.f / red[0];
  for (int c = tid; c < ncols; c += 256)
    o[c] = (bf16)(__expf(p[c] - mx) * inv);
}

__global__ __launch_bounds__(256)
void layernorm_rows_k(const float* __restrict__ in, const float* __restrict__ g,
                      const float* __restrict__ bta, bf16* __restrict__ out,
                      int ncols) {
  __shared__ float red[256];
  const int row = blockIdx.x;
  const float* p = in + (long)row * ncols;
  bf16* o = out + (long)row * ncols;
  const int tid = threadIdx.x;
  float s = 0.f;
  for (int c = tid; c < ncols; c += 256) s += p[c];
  red[tid] = s; __syncthreads();
  for (int st = 128; st > 0; st >>= 1) {
    if (tid < st) red[tid] += red[tid + st];
    __syncthreads();
  }
  float mu = red[0] / (float)ncols; __syncthreads();
  float v = 0.f;
  for (int c = tid; c < ncols; c += 256) { float d = p[c] - mu; v += d * d; }
  red[tid] = v; __syncthreads();
  for (int st = 128; st > 0; st >>= 1) {
    if (tid < st) red[tid] += red[tid + st];
    __syncthreads();
  }
  float rstd = rsqrtf(red[0] / (float)ncols + 1e-5f);
  for (int c = tid; c < ncols; c += 256)
    o[c] = (bf16)((p[c] - mu) * rstd * g[c] + bta[c]);
}

// ---------------------------------------------------------------------------
// Generic bf16 WMMA GEMM:  C[m][n] = sum_k A[m*sAm + k*sAk] * B[n*sBn + k*sBk]
// 128x64 block tile, 8 waves (4M x 2N), 32x32 per wave (4 WMMAs / K-step).
// Vectorized b128 staging when the K-stride is 1; scalar fallback otherwise.
// ---------------------------------------------------------------------------
template <bool OUT_BF16>
__global__ __launch_bounds__(256)
void wmma_gemm_k(const bf16* __restrict__ A, const bf16* __restrict__ Bm,
                 float* __restrict__ Cf, bf16* __restrict__ Cb,
                 const float* __restrict__ bias,
                 int Mdim, int Ndim, int Kdim,
                 long sAm, long sAk, long sBn, long sBk,
                 long batchA, long batchB, long batchC) {
  __shared__ __align__(64) bf16 As[128][32];
  __shared__ __align__(64) bf16 Bs[64][32];
  const int tid  = threadIdx.x;
  const int lane = tid & 31;
  const int w    = tid >> 5;
  const int wm   = w & 3;        // 0..3  (M group of 32)
  const int wn   = w >> 2;       // 0..1  (N group of 32)
  const int lh   = lane >> 4;    // lane half
  const int ln   = lane & 15;
  const int m0   = blockIdx.y * 128;
  const int n0   = blockIdx.x * 64;
  const long ofA = (long)blockIdx.z * batchA;
  const long ofB = (long)blockIdx.z * batchB;
  const long ofC = (long)blockIdx.z * batchC;

  const bool fastA = (sAk == 1) && ((sAm & 7) == 0);
  const bool fastB = (sBk == 1) && ((sBn & 7) == 0);

  v8f acc00 = {}, acc01 = {}, acc10 = {}, acc11 = {};

  for (int k0 = 0; k0 < Kdim; k0 += 32) {
    // ---- stage A tile (128 x 32) ----
    if (fastA) {
#pragma unroll
      for (int e = 0; e < 2; ++e) {
        int c  = tid + e * 256;          // 0..511, 4 chunks of 8 per row
        int mm = c >> 2, ch = (c & 3) * 8;
        int gm = m0 + mm, gk = k0 + ch;
        v8bf vv;
        if (gm < Mdim && gk + 7 < Kdim) {
          vv = *(const v8bf*)&A[ofA + (long)gm * sAm + gk];
        } else {
          vv = z8();
          if (gm < Mdim)
#pragma unroll
            for (int t = 0; t < 8; ++t)
              if (gk + t < Kdim) vv[t] = A[ofA + (long)gm * sAm + gk + t];
        }
        *(v8bf*)&As[mm][ch] = vv;
      }
    } else {
#pragma unroll
      for (int e = 0; e < 16; ++e) {
        int idx = tid + e * 256;         // 0..4095
        int mm = idx >> 5, kk = idx & 31;
        int gm = m0 + mm, gk = k0 + kk;
        bf16 av = (bf16)0.0f;
        if (gm < Mdim && gk < Kdim)
          av = A[ofA + (long)gm * sAm + (long)gk * sAk];
        As[mm][kk] = av;
      }
    }
    // ---- stage B tile (64 x 32) ----
    if (fastB) {
      {
        int c  = tid;                    // 0..255, 4 chunks of 8 per row
        int nn = c >> 2, ch = (c & 3) * 8;
        int gn = n0 + nn, gk = k0 + ch;
        v8bf vv;
        if (gn < Ndim && gk + 7 < Kdim) {
          vv = *(const v8bf*)&Bm[ofB + (long)gn * sBn + gk];
        } else {
          vv = z8();
          if (gn < Ndim)
#pragma unroll
            for (int t = 0; t < 8; ++t)
              if (gk + t < Kdim) vv[t] = Bm[ofB + (long)gn * sBn + gk + t];
        }
        *(v8bf*)&Bs[nn][ch] = vv;
      }
    } else {
#pragma unroll
      for (int e = 0; e < 8; ++e) {
        int idx = tid + e * 256;         // 0..2047
        int nn = idx >> 5, kk = idx & 31;
        int gn = n0 + nn, gk = k0 + kk;
        bf16 bv = (bf16)0.0f;
        if (gn < Ndim && gk < Kdim)
          bv = Bm[ofB + (long)gn * sBn + (long)gk * sBk];
        Bs[nn][kk] = bv;
      }
    }
    if (k0 + 32 < Kdim) {   // hint next K tile toward the caches
      __builtin_prefetch(&A[ofA + (long)(m0 + w * 16) * sAm + (long)(k0 + 32) * sAk], 0, 1);
      __builtin_prefetch(&Bm[ofB + (long)(n0 + w * 8) * sBn + (long)(k0 + 32) * sBk], 0, 1);
    }
    __syncthreads();

    // A fragments (16x32): lane<16 rows hold K 0..15, lane>=16 K 16..31
    const int ac = lh ? 16 : 0;
    const int ar0 = wm * 32 + ln;
    const int ar1 = ar0 + 16;
    v16bf a0 = cat8(*(const v8bf*)&As[ar0][ac], *(const v8bf*)&As[ar0][ac + 8]);
    v16bf a1 = cat8(*(const v8bf*)&As[ar1][ac], *(const v8bf*)&As[ar1][ac + 8]);
    // B fragments (32x16): lane holds column ln, K chunk (lh?16:0)..+15
    const int bc = lh ? 16 : 0;
    v16bf b0 = *(const v16bf*)&Bs[wn * 32 + ln][bc];
    v16bf b1 = *(const v16bf*)&Bs[wn * 32 + 16 + ln][bc];
    acc00 = wmma_bf16(a0, b0, acc00);
    acc01 = wmma_bf16(a0, b1, acc01);
    acc10 = wmma_bf16(a1, b0, acc10);
    acc11 = wmma_bf16(a1, b1, acc11);
    __syncthreads();
  }

  // C layout: vgpr r -> row (lane<16 ? r : r+8), col = ln within fragment
#pragma unroll
  for (int ai = 0; ai < 2; ++ai) {
    v8f ra0 = ai ? acc10 : acc00;
    v8f ra1 = ai ? acc11 : acc01;
#pragma unroll
    for (int r = 0; r < 8; ++r) {
      int gm = m0 + wm * 32 + ai * 16 + (lh ? r + 8 : r);
      if (gm >= Mdim) continue;
      int gn0 = n0 + wn * 32 + ln;
      int gn1 = gn0 + 16;
      float v0 = ra0[r], v1 = ra1[r];
      if (bias) { v0 += bias[gn0]; v1 += bias[gn1]; }
      long c0i = ofC + (long)gm * Ndim + gn0;
      long c1i = ofC + (long)gm * Ndim + gn1;
      if (gn0 < Ndim) { if (OUT_BF16) Cb[c0i] = (bf16)v0; else Cf[c0i] = v0; }
      if (gn1 < Ndim) { if (OUT_BF16) Cb[c1i] = (bf16)v1; else Cf[c1i] = v1; }
    }
  }
}

// ---------------------------------------------------------------------------
// Banded flash attention. One wave = 16 query rows of one (batch, head).
// Q,K,V,O are (B, S, H*HD) bf16. Head dim 64 -> 2 K-chunks of 32 per QK^T.
// ---------------------------------------------------------------------------
__global__ __launch_bounds__(256)
void flash_band_attn_k(const bf16* __restrict__ Q, const bf16* __restrict__ Kt,
                       const bf16* __restrict__ V, bf16* __restrict__ O) {
  __shared__ __align__(64) bf16 Pt[8][16][32];   // P tile per wave (16x32)
  __shared__ __align__(64) bf16 Vt[8][64][32];   // V^T tile per wave (64 cols x 32 keys)

  const int lane = threadIdx.x & 31;
  const int w    = threadIdx.x >> 5;
  const int h    = blockIdx.y;
  const int b    = blockIdx.z;
  const int q0   = blockIdx.x * 128 + w * 16;
  if (q0 >= Sc) return;                    // uniform per wave; no block barriers used
  const int lh = lane >> 4;
  const int ln = lane & 15;
  const long base = ((long)b * Sc) * (Hc * HDc) + (long)h * HDc;

  // --- Q A-fragments (cols 0..31 and 32..63) ---
  v16bf a0, a1;
  {
    int rq = q0 + ln;
    const int c0 = lh ? 16 : 0;
    if (rq < Sc) {
      const bf16* qp = Q + base + (long)rq * (Hc * HDc);
      a0 = cat8(*(const v8bf*)(qp + c0),      *(const v8bf*)(qp + c0 + 8));
      a1 = cat8(*(const v8bf*)(qp + 32 + c0), *(const v8bf*)(qp + 32 + c0 + 8));
    } else { a0 = z16(); a1 = z16(); }
  }

  v8f o0 = {}, o1 = {}, o2 = {}, o3 = {};
  float mrun[8], lrun[8];
#pragma unroll
  for (int r = 0; r < 8; ++r) { mrun[r] = -3.0e38f; lrun[r] = 0.f; }

  int jLo = (q0 > (WINc - 1)) ? ((q0 - (WINc - 1)) & ~31) : 0;
  int jHi = q0 + 16 + (WINc - 1);
  if (jHi > Sc) jHi = Sc;
  const float scl = 0.125f;                // 1/sqrt(64)

  for (int j0 = jLo; j0 < jHi; j0 += 32) {
    // ---- scores: two 16-key fragments, each K-dim 64 via 2 chained WMMAs ----
    v8f sc0 = {}, sc1 = {};
    {
      const int c0 = lh ? 16 : 0;
      int jk0 = j0 + ln, jk1 = j0 + 16 + ln;
      const bf16* kp0 = Kt + base + (long)jk0 * (Hc * HDc);
      const bf16* kp1 = Kt + base + (long)jk1 * (Hc * HDc);
      v16bf bk;
      bk  = (jk0 < Sc) ? *(const v16bf*)(kp0 + c0)      : z16();
      sc0 = wmma_bf16(a0, bk, sc0);
      bk  = (jk0 < Sc) ? *(const v16bf*)(kp0 + 32 + c0) : z16();
      sc0 = wmma_bf16(a1, bk, sc0);
      bk  = (jk1 < Sc) ? *(const v16bf*)(kp1 + c0)      : z16();
      sc1 = wmma_bf16(a0, bk, sc1);
      bk  = (jk1 < Sc) ? *(const v16bf*)(kp1 + 32 + c0) : z16();
      sc1 = wmma_bf16(a1, bk, sc1);
    }

    // ---- band mask + online softmax (per C-fragment row) ----
    float p0a[8], p1a[8], alpha[8];
#pragma unroll
    for (int r = 0; r < 8; ++r) {
      int i   = q0 + (lh ? r + 8 : r);
      int jj0 = j0 + ln;
      int jj1 = j0 + 16 + ln;
      int d0 = i - jj0; if (d0 < 0) d0 = -d0;
      int d1 = i - jj1; if (d1 < 0) d1 = -d1;
      bool ok0 = (i < Sc) && (jj0 < Sc) && (d0 < WINc);
      bool ok1 = (i < Sc) && (jj1 < Sc) && (d1 < WINc);
      float s0 = ok0 ? sc0[r] * scl : -3.0e38f;
      float s1 = ok1 ? sc1[r] * scl : -3.0e38f;
      float tmax = fmaxf(s0, s1);
      tmax = fmaxf(tmax, __shfl_xor(tmax, 1));
      tmax = fmaxf(tmax, __shfl_xor(tmax, 2));
      tmax = fmaxf(tmax, __shfl_xor(tmax, 4));
      tmax = fmaxf(tmax, __shfl_xor(tmax, 8));
      float mnew = fmaxf(mrun[r], tmax);
      float al = __expf(mrun[r] - mnew);
      float p0 = ok0 ? __expf(s0 - mnew) : 0.f;   // explicit zero avoids mask bug
      float p1 = ok1 ? __expf(s1 - mnew) : 0.f;
      float ps = p0 + p1;
      ps += __shfl_xor(ps, 1);
      ps += __shfl_xor(ps, 2);
      ps += __shfl_xor(ps, 4);
      ps += __shfl_xor(ps, 8);
      lrun[r] = lrun[r] * al + ps;
      mrun[r] = mnew;
      alpha[r] = al;
      p0a[r] = p0; p1a[r] = p1;
    }
#pragma unroll
    for (int r = 0; r < 8; ++r) {
      o0[r] *= alpha[r]; o1[r] *= alpha[r]; o2[r] *= alpha[r]; o3[r] *= alpha[r];
    }

    // ---- P (C layout) -> LDS, V tile -> LDS transposed ----
#pragma unroll
    for (int r = 0; r < 8; ++r) {
      int rr = lh ? r + 8 : r;
      Pt[w][rr][ln]      = (bf16)p0a[r];
      Pt[w][rr][16 + ln] = (bf16)p1a[r];
    }
    {
      int jv = j0 + lane;            // each lane stages one key row of V
      if (jv < Sc) {
        const bf16* vp = V + base + (long)jv * (Hc * HDc);
#pragma unroll
        for (int cb = 0; cb < 8; ++cb) {
          v8bf vv = *(const v8bf*)(vp + cb * 8);
#pragma unroll
          for (int t = 0; t < 8; ++t) Vt[w][cb * 8 + t][lane] = vv[t];
        }
      } else {
#pragma unroll
        for (int c = 0; c < 64; ++c) Vt[w][c][lane] = (bf16)0.0f;
      }
    }
    asm volatile("s_wait_dscnt 0" ::: "memory");   // same-wave LDS RAW ordering

    // ---- reload P as A fragment, V columns as B fragments; O += P @ V ----
    v16bf ap;
    {
      const int c0 = lh ? 16 : 0;
      ap = cat8(*(const v8bf*)&Pt[w][ln][c0], *(const v8bf*)&Pt[w][ln][c0 + 8]);
    }
    {
      const int c0 = lh ? 16 : 0;
      v16bf bv;
      bv = *(const v16bf*)&Vt[w][0 * 16 + ln][c0];  o0 = wmma_bf16(ap, bv, o0);
      bv = *(const v16bf*)&Vt[w][1 * 16 + ln][c0];  o1 = wmma_bf16(ap, bv, o1);
      bv = *(const v16bf*)&Vt[w][2 * 16 + ln][c0];  o2 = wmma_bf16(ap, bv, o2);
      bv = *(const v16bf*)&Vt[w][3 * 16 + ln][c0];  o3 = wmma_bf16(ap, bv, o3);
    }
  }

  // ---- normalize + store (diagonal always in band => lrun > 0) ----
#pragma unroll
  for (int r = 0; r < 8; ++r) {
    int i = q0 + (lh ? r + 8 : r);
    if (i >= Sc) continue;
    float inv = 1.f / lrun[r];
    bf16* op = O + base + (long)i * (Hc * HDc);
    op[0 * 16 + ln] = (bf16)(o0[r] * inv);
    op[1 * 16 + ln] = (bf16)(o1[r] * inv);
    op[2 * 16 + ln] = (bf16)(o2[r] * inv);
    op[3 * 16 + ln] = (bf16)(o3[r] * inv);
  }
}

// ---------------------------------------------------------------------------
// Host orchestration
// ---------------------------------------------------------------------------
static inline void* bump(char*& p, size_t bytes) {
  void* r = (void*)p;
  p += (bytes + 255) & ~(size_t)255;
  return r;
}

extern "C" void kernel_launch(void* const* d_in, const int* in_sizes, int n_in,
                              void* d_out, int out_size, void* d_ws,
                              size_t ws_size, hipStream_t stream) {
  (void)in_sizes; (void)n_in; (void)out_size; (void)ws_size;

  const float* x    = (const float*)d_in[0];
  const float* pmem = (const float*)d_in[1];
  const float* mkey = (const float*)d_in[2];
  const float* mWk  = (const float*)d_in[3];
  const float* mWv  = (const float*)d_in[4];
  const float* mWq  = (const float*)d_in[5];
  const float* aWq  = (const float*)d_in[6];
  const float* aWk  = (const float*)d_in[7];
  const float* aWv  = (const float*)d_in[8];
  const float* aWo  = (const float*)d_in[9];
  const float* ln1g = (const float*)d_in[10];
  const float* ln1b = (const float*)d_in[11];
  const float* ln2g = (const float*)d_in[12];
  const float* ln2b = (const float*)d_in[13];
  const float* outW = (const float*)d_in[14];
  const float* outb = (const float*)d_in[15];
  float* out = (float*)d_out;

  char* p = (char*)d_ws;
  bf16* comb   = (bf16*)bump(p, (size_t)BSc * Dc * 2);
  bf16* wkb    = (bf16*)bump(p, (size_t)Dc * Dc * 2);
  bf16* wvb    = (bf16*)bump(p, (size_t)Dc * Dc * 2);
  bf16* wqb    = (bf16*)bump(p, (size_t)Dc * Dc * 2);
  bf16* mkb    = (bf16*)bump(p, (size_t)Mc * Dc * 2);
  bf16* aWqb   = (bf16*)bump(p, (size_t)Dc * Dc * 2);
  bf16* aWkb   = (bf16*)bump(p, (size_t)Dc * Dc * 2);
  bf16* aWvb   = (bf16*)bump(p, (size_t)Dc * Dc * 2);
  bf16* aWob   = (bf16*)bump(p, (size_t)Dc * Dc * 2);
  bf16* outWb  = (bf16*)bump(p, (size_t)Dc * Dc * 2);
  bf16* kb     = (bf16*)bump(p, (size_t)BSc * Dc * 2);
  bf16* vb     = (bf16*)bump(p, (size_t)BSc * Dc * 2);
  bf16* qb     = (bf16*)bump(p, (size_t)BSc * Dc * 2);
  float* logit = (float*)bump(p, (size_t)BSc * Mc * 4);
  bf16* wwb    = (bf16*)bump(p, (size_t)BSc * Mc * 2);
  bf16* wrb    = (bf16*)bump(p, (size_t)BSc * Mc * 2);
  bf16* stateb = (bf16*)bump(p, (size_t)Bc * Mc * Dc * 2);
  float* memof = (float*)bump(p, (size_t)BSc * Dc * 4);
  bf16* hb     = (bf16*)bump(p, (size_t)BSc * Dc * 2);
  bf16* qhb    = (bf16*)bump(p, (size_t)BSc * Dc * 2);
  bf16* khb    = (bf16*)bump(p, (size_t)BSc * Dc * 2);
  bf16* vhb    = (bf16*)bump(p, (size_t)BSc * Dc * 2);
  bf16* aob    = (bf16*)bump(p, (size_t)BSc * Dc * 2);
  float* ao2f  = (float*)bump(p, (size_t)BSc * Dc * 4);
  bf16* h2b    = (bf16*)bump(p, (size_t)BSc * Dc * 2);

  auto cvt = [&](const float* src, bf16* dst, int n) {
    f32_to_bf16_k<<<(n + 255) / 256, 256, 0, stream>>>(src, dst, n);
  };
  auto gemmB = [&](const bf16* A, const bf16* Bm, bf16* Cb,
                   int Md, int Nd, int Kd, long sAm, long sAk, long sBn,
                   long sBk, long bA, long bB, long bC, int nb) {
    dim3 g((Nd + 63) / 64, (Md + 127) / 128, nb);
    wmma_gemm_k<true><<<g, 256, 0, stream>>>(A, Bm, nullptr, Cb, nullptr,
                                             Md, Nd, Kd, sAm, sAk, sBn, sBk,
                                             bA, bB, bC);
  };
  auto gemmF = [&](const bf16* A, const bf16* Bm, float* Cf, const float* bias,
                   int Md, int Nd, int Kd, long sAm, long sAk, long sBn,
                   long sBk, long bA, long bB, long bC, int nb) {
    dim3 g((Nd + 63) / 64, (Md + 127) / 128, nb);
    wmma_gemm_k<false><<<g, 256, 0, stream>>>(A, Bm, Cf, nullptr, bias,
                                              Md, Nd, Kd, sAm, sAk, sBn, sBk,
                                              bA, bB, bC);
  };

  // --- prep: combined + bf16 weights ---
  build_combined_k<<<(BSc * Dc + 255) / 256, 256, 0, stream>>>(x, pmem, comb);
  cvt(mWk, wkb, Dc * Dc);   cvt(mWv, wvb, Dc * Dc);   cvt(mWq, wqb, Dc * Dc);
  cvt(mkey, mkb, Mc * Dc);
  cvt(aWq, aWqb, Dc * Dc);  cvt(aWk, aWkb, Dc * Dc);
  cvt(aWv, aWvb, Dc * Dc);  cvt(aWo, aWob, Dc * Dc);
  cvt(outW, outWb, Dc * Dc);

  // --- neural memory: k/v/q = combined @ W  (B[n][k] = W[k][n]) ---
  gemmB(comb, wkb, kb, BSc, Dc, Dc, Dc, 1, 1, Dc, 0, 0, 0, 1);
  gemmB(comb, wvb, vb, BSc, Dc, Dc, Dc, 1, 1, Dc, 0, 0, 0, 1);
  gemmB(comb, wqb, qb, BSc, Dc, Dc, Dc, 1, 1, Dc, 0, 0, 0, 1);

  // w_write = softmax(k @ mem_keys^T)   (B[n][k] = mem_keys[n][k], fastB path)
  gemmF(kb, mkb, logit, nullptr, BSc, Mc, Dc, Dc, 1, Dc, 1, 0, 0, 0, 1);
  softmax_rows_k<<<BSc, 256, 0, stream>>>(logit, wwb, Mc);
  // w_read = softmax(q @ mem_keys^T)
  gemmF(qb, mkb, logit, nullptr, BSc, Mc, Dc, Dc, 1, Dc, 1, 0, 0, 0, 1);
  softmax_rows_k<<<BSc, 256, 0, stream>>>(logit, wrb, Mc);

  // state[b] = w_write[b]^T @ v[b] : A[m][s]=ww[s*Mc+m], B[d][s]=v[s*Dc+d]
  gemmB(wwb, vb, stateb, Mc, Dc, Sc,
        /*sAm=*/1, /*sAk=*/Mc, /*sBn=*/1, /*sBk=*/Dc,
        (long)Sc * Mc, (long)Sc * Dc, (long)Mc * Dc, Bc);

  // mem_out[b] = w_read[b] @ state[b] : B[d][m]=state[m*Dc+d]
  gemmF(wrb, stateb, memof, nullptr, Sc, Dc, Mc,
        /*sAm=*/Mc, /*sAk=*/1, /*sBn=*/1, /*sBk=*/Dc,
        (long)Sc * Mc, (long)Mc * Dc, (long)Sc * Dc, Bc);

  // h = LN1(mem_out)
  layernorm_rows_k<<<BSc, 256, 0, stream>>>(memof, ln1g, ln1b, hb, Dc);

  // attention projections
  gemmB(hb, aWqb, qhb, BSc, Dc, Dc, Dc, 1, 1, Dc, 0, 0, 0, 1);
  gemmB(hb, aWkb, khb, BSc, Dc, Dc, Dc, 1, 1, Dc, 0, 0, 0, 1);
  gemmB(hb, aWvb, vhb, BSc, Dc, Dc, Dc, 1, 1, Dc, 0, 0, 0, 1);

  // banded flash attention
  {
    dim3 g((Sc + 127) / 128, Hc, Bc);
    flash_band_attn_k<<<g, 256, 0, stream>>>(qhb, khb, vhb, aob);
  }

  // ao = attn_out @ attn_Wo, LN2, final projection + bias -> d_out
  gemmF(aob, aWob, ao2f, nullptr, BSc, Dc, Dc, Dc, 1, 1, Dc, 0, 0, 0, 1);
  layernorm_rows_k<<<BSc, 256, 0, stream>>>(ao2f, ln2g, ln2b, h2b, Dc);
  gemmF(h2b, outWb, out, outb, BSc, Dc, Dc, Dc, 1, 1, Dc, 0, 0, 0, 1);
}
